// ModifiedInnerShiftTripleFunction_25864293056523
// MI455X (gfx1250) — matrix operand order repaired and checked by reading, not compiled, not run on hardware
//
#include <hip/hip_runtime.h>
#include <hip/hip_bf16.h>

typedef __attribute__((ext_vector_type(16))) _Float16 v16h;
typedef __attribute__((ext_vector_type(8)))  _Float16 v8h;
typedef __attribute__((ext_vector_type(8)))  float    v8f;

#define BZ 8
#define C_IN 512
#define CH 256
#define S 4096              // 64*64 spatial
#define QTILE_BLOCK 128     // q rows per block (8 waves x 16)
#define NTILES (S / 16)     // 256 k-tiles
#define LDS_ROW_BYTES 528   // 256 f16 (512B) + 16B pad to break LDS bank conflicts

#if defined(__has_builtin)
#if __has_builtin(__builtin_amdgcn_sched_group_barrier)
#define HAVE_SCHED_GB 1
#endif
#endif

// ---------------- kernel 1: out[:, 0:512] = input (pure bandwidth, float4) ----
__global__ void copy_fl_kernel(const float* __restrict__ in, float* __restrict__ out) {
    size_t i = (size_t)blockIdx.x * blockDim.x + threadIdx.x;      // float4 index
    size_t n4 = (size_t)BZ * C_IN * S / 4;
    if (i >= n4) return;
    size_t e = i * 4;
    int b = (int)(e / ((size_t)C_IN * S));
    size_t r = e - (size_t)b * C_IN * S;                           // c*S + s
    float4 v = *(const float4*)(in + (size_t)b * C_IN * S + r);
    *(float4*)(out + (size_t)b * 768 * S + r) = v;
}

// ---------------- kernel 2: rnorm[b*S+k] = 1 / max(||L[b,:,k]||, 1e-12) -------
__global__ void rnorm_kernel(const float* __restrict__ in, float* __restrict__ rnorm) {
    int t = blockIdx.x * blockDim.x + threadIdx.x;                 // 0 .. BZ*S-1
    int b = t >> 12;
    int k = t & (S - 1);
    const float* p = in + ((size_t)(b * C_IN + CH)) * S + k;       // latter[b][c][k]
    float s = 0.f;
#pragma unroll 8
    for (int c = 0; c < CH; ++c) {
        float v = p[(size_t)c * S];
        s += v * v;
    }
    rnorm[t] = 1.0f / fmaxf(sqrtf(s), 1e-12f);
}

// ---------------- kernel 3: streamed WMMA sim + masked argmax + gather --------
__global__ __launch_bounds__(256, 1)
void argmax_shift_kernel(const float* __restrict__ in, const int* __restrict__ flag,
                         const float* __restrict__ rnorm, float* __restrict__ out) {
    __shared__ __align__(16) unsigned char ldsB[2][16 * LDS_ROW_BYTES]; // double-buffered B tile (f16)
    __shared__ int idxshare[QTILE_BLOCK];

    const int b    = blockIdx.y;
    const int qblk = blockIdx.x * QTILE_BLOCK;
    const int tid  = threadIdx.x;
    const int wave = tid >> 5;
    const int lane = tid & 31;
    const int m16  = lane & 15;
    const int hi   = lane >> 4;
    const int q0   = qblk + wave * 16;

    const float* former = in + (size_t)b * C_IN * S;   // [c][s], c in [0,256)
    const float* latter = former + (size_t)CH * S;     // [c][s], c in [0,256)
    const float* rn     = rnorm + (size_t)b * S;

    // ---- Load A tile once: 16 q-rows x 256 ch, f32 -> f16, WMMA A layout ----
    // lanes 0-15 (row M=m16): elems 0..7 = K 0..7,  elems 8..15 = K 16..23 (per 32-K step)
    // lanes 16-31 (row M=m16): elems 0..7 = K 8..15, elems 8..15 = K 24..31
    v16h A[8];
    {
        const float* fq = former + (q0 + m16);         // + c*S per channel
#pragma unroll
        for (int kk = 0; kk < 8; ++kk) {
#pragma unroll
            for (int i = 0; i < 8; ++i) {
                float v0 = fq[(size_t)(kk * 32 + hi * 8 + i) * S];
                float v1 = fq[(size_t)(kk * 32 + 16 + hi * 8 + i) * S];
                A[kk][i]     = (_Float16)v0;
                A[kk][i + 8] = (_Float16)v1;
            }
        }
    }

    // staging-thread assignment: 16 k-rows x (16 threads x 16 channels)
    const int sr  = tid & 15;          // k row within tile
    const int sc0 = (tid >> 4) * 16;   // channel start
    const float* lbase = latter + (size_t)sc0 * S;

    float bestv[8];
    int   besti[8];
#pragma unroll
    for (int m = 0; m < 8; ++m) { bestv[m] = -3.0e38f; besti[m] = 0; }

    // ---- software pipeline: prefetch raw f32 B tile 0 into registers ----
    float pre[16];
    float prescale;
    int   prefl;
    {
        const float* lp = lbase + sr;                  // k = 0 + sr
        prescale = rn[sr];
        prefl    = flag[m16];                          // flag for this lane's column, tile 0
#pragma unroll
        for (int i = 0; i < 16; ++i) pre[i] = lp[(size_t)i * S];
    }

    for (int t = 0; t < NTILES; ++t) {
        const int k0  = t * 16;
        const int buf = t & 1;

        // ---- convert + store prefetched tile t into LDS (scaled, f16) ----
        // All load results of tile t are consumed HERE (before the t+1 prefetch is
        // issued), so the only wait the barrier needs is dscnt -> the t+1 loads
        // stay outstanding across the barrier + WMMA block.
        {
            v8h h0, h1;
#pragma unroll
            for (int i = 0; i < 8; ++i) h0[i] = (_Float16)(pre[i] * prescale);
#pragma unroll
            for (int i = 0; i < 8; ++i) h1[i] = (_Float16)(pre[i + 8] * prescale);
            v8h* dst = (v8h*)&ldsB[buf][sr * LDS_ROW_BYTES + sc0 * 2];
            dst[0] = h0;
            dst[1] = h1;
        }
        // branch-free column mask: masked columns get -3e38 bias -> can never beat
        // the -3e38 init (strict >), exactly like the reference's -1e30 fill.
        const float vbias = (prefl == 0) ? 0.f : -3.0e38f;

        // ---- issue global prefetch for tile t+1 (overlaps barrier + WMMA) ----
        {
            const int tn = (t + 1 < NTILES) ? (t + 1) : t;   // uniform clamp
            const int kn = tn * 16 + sr;
            prescale = rn[kn];
            prefl    = flag[tn * 16 + m16];
            const float* lp = lbase + kn;
#pragma unroll
            for (int i = 0; i < 16; ++i) pre[i] = lp[(size_t)i * S];
        }

        // one barrier per tile: double buffering covers the write-after-compute hazard
        __syncthreads();

        // ---- load all 8 B operands, then 8 back-to-back WMMAs (K = 256) ----
        // B layout: lane n<16 holds column N=n, elems 0..7 = K 0..7, elems 8..15 = K 16..23; lanes 16-31 hold K 8..15, K 24..31
        const unsigned char* src = &ldsB[buf][m16 * LDS_ROW_BYTES + hi * 16];
        v16h Bv[8];
#pragma unroll
        for (int kk = 0; kk < 8; ++kk) {
            const v8h* sp = (const v8h*)(src + kk * 64);
            v8h lo = sp[0], hi8 = sp[2];
#pragma unroll
            for (int i = 0; i < 8; ++i) { Bv[kk][i] = lo[i]; Bv[kk][i + 8] = hi8[i]; }
        }
        v8f acc = {0.f, 0.f, 0.f, 0.f, 0.f, 0.f, 0.f, 0.f};
#pragma unroll
        for (int kk = 0; kk < 8; ++kk) {
            acc = __builtin_amdgcn_wmma_f32_16x16x32_f16(
                false, A[kk], false, Bv[kk], (short)0, acc, false, false);
        }
#ifdef HAVE_SCHED_GB
        // pin the pipeline: 16 LDS reads first, then the 8 WMMAs -> partial dscnt
        // waits instead of a full s_wait_dscnt 0 before every WMMA
        __builtin_amdgcn_sched_group_barrier(0x100, 16, 0);  // DS read x16
        __builtin_amdgcn_sched_group_barrier(0x008, 8, 0);   // WMMA    x8
#endif

        // ---- masked running argmax (branch-free; lane owns column k = k0+m16) ----
        // C/D layout: lanes 0-15 hold M=0..7 in acc[0..7]; lanes 16-31 hold M=8..15
#pragma unroll
        for (int m = 0; m < 8; ++m) {
            float cv = acc[m] + vbias;
            if (cv > bestv[m]) { bestv[m] = cv; besti[m] = k0 + m16; }
        }
    }

    // ---- argmax reduction across the 16 lanes sharing each M row ----
    // strictly-greater wins; on exact tie keep smaller k (first occurrence, like jnp.argmax)
#pragma unroll
    for (int m = 0; m < 8; ++m) {
#pragma unroll
        for (int sft = 1; sft < 16; sft <<= 1) {
            float ov = __shfl_xor(bestv[m], sft, 32);
            int   oi = __shfl_xor(besti[m], sft, 32);
            if (ov > bestv[m] || (ov == bestv[m] && oi < besti[m])) {
                bestv[m] = ov; besti[m] = oi;
            }
        }
    }
    if (m16 == 0) {            // lane 0 (M rows 0..7) and lane 16 (M rows 8..15)
#pragma unroll
        for (int m = 0; m < 8; ++m) idxshare[wave * 16 + hi * 8 + m] = besti[m];
    }
    __syncthreads();

    // ---- gather + write shifted block: out[b, 512+c, q] ----
    const int row  = tid & 127;        // q row within block -> coalesced on q
    const int csel = tid >> 7;         // channel half
    const int q    = qblk + row;
    const int widx = idxshare[row];
    const int flq  = flag[q];
    float* outs = out + ((size_t)b * 768 + 512) * S;
#pragma unroll 4
    for (int c = csel * 128; c < csel * 128 + 128; ++c) {
        float v = 0.f;
        if (flq == 1) v = latter[(size_t)c * S + widx];
        outs[(size_t)c * S + q] = v;
    }
}

extern "C" void kernel_launch(void* const* d_in, const int* in_sizes, int n_in,
                              void* d_out, int out_size, void* d_ws, size_t ws_size,
                              hipStream_t stream) {
    const float* x    = (const float*)d_in[0];
    const int*   flag = (const int*)d_in[2];   // d_in[1] = 2D mask (same data, unused)
    float*       out  = (float*)d_out;
    float*       rnorm = (float*)d_ws;         // 8*4096 floats = 128 KB scratch

    // 1) copy former+latter -> out[:, 0:512]
    {
        size_t n4 = (size_t)BZ * C_IN * S / 4;
        int blocks = (int)((n4 + 255) / 256);
        copy_fl_kernel<<<blocks, 256, 0, stream>>>(x, out);
    }
    // 2) inverse row norms of latter
    rnorm_kernel<<<(BZ * S) / 256, 256, 0, stream>>>(x, rnorm);
    // 3) streamed WMMA sim + masked argmax + gather-write of shifted block
    dim3 grid(S / QTILE_BLOCK, BZ);
    argmax_shift_kernel<<<grid, 256, 0, stream>>>(x, flag, rnorm, out);
}
